// update_e_spherenet_48034914238946
// MI455X (gfx1250) — matrix operand order repaired and checked by reading, not compiled, not run on hardware
//
#include <hip/hip_runtime.h>

// ---------------- problem constants ----------------
#define E_N  200000
#define T_N  600000
#define Hd   128
#define Id   64
#define NRr  6
#define SBFD 42       // 7*6
#define TTD  294      // 7*7*6
#define TTP  320      // TTD padded to K-multiple of 32

// ---------------- WMMA types ----------------
typedef __attribute__((ext_vector_type(16))) __bf16 v16bf;
typedef __attribute__((ext_vector_type(8)))  float  v8f;

union BF16x16 { v16bf v; unsigned short s[16]; uint4 q[2]; };
union F32x8   { v8f v; float f[8]; };

__device__ __forceinline__ unsigned short f2bf(float f) {
  union { float f; unsigned u; } x; x.f = f;
  unsigned r = x.u + 0x7FFFu + ((x.u >> 16) & 1u);   // round-to-nearest-even
  return (unsigned short)(r >> 16);
}
__device__ __forceinline__ float silu(float x) { return x / (1.0f + __expf(-x)); }

__device__ __forceinline__ v8f wmma_bf16(v16bf a, v16bf b, v8f c) {
  // (neg_a, A, neg_b, B, c_mod, C, reuse_a, reuse_b)
  return __builtin_amdgcn_wmma_f32_16x16x32_bf16(false, a, false, b, (short)0, c, false, false);
}

// A fragment (16x32 bf16) from row-major fp32, ISA layout (CDNA5 ISA 7.12.2):
//   lane<16: M=lane, K chunks [kbase+0..7] and [kbase+16..23]
//   lane>=16: M=lane-16, K chunks [kbase+8..15] and [kbase+24..31]
__device__ __forceinline__ v16bf afrag_f32(const float* X, int ld, int kbase, int lane) {
  int m = lane & 15, hi = lane >> 4;
  const float* r = X + (size_t)m * ld + kbase + hi * 8;
  BF16x16 a;
#pragma unroll
  for (int j = 0; j < 8; ++j) a.s[j] = f2bf(r[j]);
#pragma unroll
  for (int j = 0; j < 8; ++j) a.s[8 + j] = f2bf(r[16 + j]);
  return a.v;
}

// A fragment from bf16 LDS staging (two 16B ds reads per lane)
__device__ __forceinline__ v16bf afrag_lds(const unsigned short* sh, int ld, int kbase, int lane) {
  int m = lane & 15, hi = lane >> 4;
  const unsigned short* r = sh + m * ld + kbase + hi * 8;
  BF16x16 a;
  a.q[0] = *(const uint4*)(r);
  a.q[1] = *(const uint4*)(r + 16);
  return a.v;
}

// B fragment from pre-swizzled weight store: one contiguous 32B load per lane
__device__ __forceinline__ v16bf bfrag(const unsigned short* wf, int NT, int kt, int nt, int lane) {
  return *(const v16bf*)(wf + (((size_t)(kt * NT + nt) * 32) + lane) * 16);
}

// LDS offset (addrspace(3) value) for async-to-LDS destinations
__device__ __forceinline__ unsigned lds_off(const void* p) {
  return (unsigned)(uintptr_t)(__attribute__((address_space(3))) const void*)p;
}

// ---------------- fragment-weight store layout (ushort offsets) ----------------
#define WF_JI   0
#define WF_KJ   16384
#define WF_DN   32768
#define WF_UP   40960
#define WF_B1   49152
#define WF_B2   65536
#define WF_LIN  81920
#define WF_A1_0 98304
#define WF_A1_1 114688
#define WF_A2_0 131072
#define WF_A2_1 147456
#define WF_T1P  163840   // W_t1  padded [320,16]  -> 5120
#define WF_S1P  168960   // W_sbf1 padded [64,16]  -> 1024
#define WF_T2P  169984   // W_t2  padded [32,64]   -> 2048
#define WF_S2P  172032   // W_sbf2 padded [32,64]  -> 2048
#define WF_TOT  174080

// ================= kernel 0: weight prep (bf16 frag swizzle + W_r fold) =================
__global__ void kprep(const float* Wji, const float* Wkj, const float* Wdn, const float* Wup,
                      const float* Wb1, const float* Wb2, const float* Wlin,
                      const float* Wa1, const float* Wa2,
                      const float* Wr1, const float* Wr2,
                      const float* Ws1, const float* Ws2, const float* Wt1, const float* Wt2,
                      unsigned short* wf, float* wr) {
  int id = blockIdx.x * blockDim.x + threadIdx.x;
  if (id < WF_TOT) {
    const int cum[16] = {WF_JI, WF_KJ, WF_DN, WF_UP, WF_B1, WF_B2, WF_LIN,
                         WF_A1_0, WF_A1_1, WF_A2_0, WF_A2_1,
                         WF_T1P, WF_S1P, WF_T2P, WF_S2P, WF_TOT};
    const short padN[15] = {128,128, 64,128,128,128,128,128,128,128,128, 16, 16, 64, 64};
    const short srcK[15] = {128,128,128, 64,128,128,128,128,128,128,128,294, 42,  8,  8};
    const short srcN[15] = {128,128, 64,128,128,128,128,128,128,128,128,  8,  8, 64, 64};
    int mi = 0;
    while (id >= cum[mi + 1]) mi++;
    int local = id - cum[mi];
    const float* src;
    switch (mi) {
      case 0:  src = Wji;         break;
      case 1:  src = Wkj;         break;
      case 2:  src = Wdn;         break;
      case 3:  src = Wup;         break;
      case 4:  src = Wb1;         break;
      case 5:  src = Wb2;         break;
      case 6:  src = Wlin;        break;
      case 7:  src = Wa1;         break;
      case 8:  src = Wa1 + 16384; break;
      case 9:  src = Wa2;         break;
      case 10: src = Wa2 + 16384; break;
      case 11: src = Wt1;         break;
      case 12: src = Ws1;         break;
      case 13: src = Wt2;         break;
      default: src = Ws2;         break;
    }
    int NT   = padN[mi] / 16;
    int j    = local & 15;
    int lane = (local >> 4) & 31;
    int rest = local >> 9;
    int nt   = rest % NT;
    int kt   = rest / NT;
    int row  = kt * 32 + (lane >> 4) * 8 + (j & 7) + (j >> 3) * 16;
    int col  = nt * 16 + (lane & 15);
    float v  = (row < srcK[mi] && col < srcN[mi]) ? src[(size_t)row * srcN[mi] + col] : 0.0f;
    wf[id] = f2bf(v);
  } else if (id < WF_TOT + NRr * Hd) {
    int l = id - WF_TOT;
    int q = l / Hd, col = l % Hd;
    float s = 0.f;
#pragma unroll
    for (int p = 0; p < 8; ++p) s += Wr1[q * 8 + p] * Wr2[p * Hd + col];
    wr[l] = s;  // W_r = W_rbf1 @ W_rbf2, [6,128]
  }
}

// ================= kernel 1: zero segment accumulator =================
__global__ void kzero(float* p, long long n) {
  long long i = (long long)blockIdx.x * blockDim.x + threadIdx.x;
  long long stride = (long long)gridDim.x * blockDim.x;
  for (; i < n; i += stride) p[i] = 0.0f;
}

// ================= kernel 2: edge phase 1 (x_ji, x_kj*rbf_h, down-proj) =================
__global__ void kedge1(const float* __restrict__ e1, const float* __restrict__ rbf,
                       const float* __restrict__ b_ji, const float* __restrict__ b_kj,
                       const unsigned short* __restrict__ wf, const float* __restrict__ wr,
                       float* __restrict__ xji, float* __restrict__ xkjd) {
  __shared__ __align__(16) unsigned short sh_x[16 * Hd];
  __shared__ float sh_rbf[16 * 8];
  int lane = threadIdx.x;
  int row0 = blockIdx.x * 16;
  int n = lane & 15, hi = lane >> 4;

  if (lane < 16) {
#pragma unroll
    for (int q = 0; q < NRr; ++q) sh_rbf[lane * 8 + q] = rbf[(size_t)(row0 + lane) * NRr + q];
  }
  __syncthreads();

  v16bf A[4];
#pragma unroll
  for (int kt = 0; kt < 4; ++kt) A[kt] = afrag_f32(e1 + (size_t)row0 * Hd, Hd, kt * 32, lane);

  for (int nt = 0; nt < 8; ++nt) {
    int col = nt * 16 + n;
    float bj = b_ji[col], bk = b_kj[col];
    F32x8 cj, ck;
#pragma unroll
    for (int g = 0; g < 8; ++g) { cj.f[g] = bj; ck.f[g] = bk; }
#pragma unroll
    for (int kt = 0; kt < 4; ++kt) {
      cj.v = wmma_bf16(A[kt], bfrag(wf + WF_JI, 8, kt, nt, lane), cj.v);
      ck.v = wmma_bf16(A[kt], bfrag(wf + WF_KJ, 8, kt, nt, lane), ck.v);
    }
    float wrc[NRr];
#pragma unroll
    for (int q = 0; q < NRr; ++q) wrc[q] = wr[q * Hd + col];
#pragma unroll
    for (int g = 0; g < 8; ++g) {
      int r = g + hi * 8;
      xji[(size_t)(row0 + r) * Hd + col] = silu(cj.f[g]);
      float rh = 0.f;
#pragma unroll
      for (int q = 0; q < NRr; ++q) rh += sh_rbf[r * 8 + q] * wrc[q];
      sh_x[r * Hd + col] = f2bf(silu(ck.f[g]) * rh);
    }
  }
  __syncthreads();

  v16bf AK[4];
#pragma unroll
  for (int kt = 0; kt < 4; ++kt) AK[kt] = afrag_lds(sh_x, Hd, kt * 32, lane);
  for (int nt = 0; nt < 4; ++nt) {
    F32x8 c;
#pragma unroll
    for (int g = 0; g < 8; ++g) c.f[g] = 0.f;
#pragma unroll
    for (int kt = 0; kt < 4; ++kt)
      c.v = wmma_bf16(AK[kt], bfrag(wf + WF_DN, 4, kt, nt, lane), c.v);
#pragma unroll
    for (int g = 0; g < 8; ++g)
      xkjd[(size_t)(row0 + g + hi * 8) * Id + nt * 16 + n] = silu(c.f[g]);
  }
}

// ================= kernel 3: triplet phase (WMMA basis proj + gather + scatter-add) ========
__global__ void ktrip(const float* __restrict__ sbf, const float* __restrict__ t,
                      const int* __restrict__ idx_kj, const int* __restrict__ idx_ji,
                      const unsigned short* __restrict__ wf,
                      const float* __restrict__ xkjd, float* __restrict__ acc) {
  __shared__ __align__(16) float          sh_traw[16 * TTD];   // raw fp32 t tile (async target)
  __shared__ __align__(16) unsigned short sh_t[16 * TTP];      // bf16, K-padded
  __shared__ __align__(16) unsigned short sh_s[16 * 64];       // bf16 sbf, K-padded
  __shared__ __align__(16) unsigned short sh_mt[16 * 32];      // mid_t, K-padded to 32
  __shared__ __align__(16) unsigned short sh_ms[16 * 32];      // mid_s
  __shared__ int sh_kj[16], sh_ji[16];
  int lane = threadIdx.x;
  int row0 = blockIdx.x * 16;
  int n = lane & 15, hi = lane >> 4;

  // --- async-stage the 18816B t tile into LDS (ASYNCcnt path) ---
  {
    const float* tg = t + (size_t)row0 * TTD;   // 16-byte aligned: 16*294*4 = 18816 per tile
    for (int i = lane * 4; i < 16 * TTD; i += 32 * 4) {
      unsigned dst = lds_off(&sh_traw[i]);
      const float* g = tg + i;
      asm volatile("global_load_async_to_lds_b128 %0, %1, off"
                   :: "v"(dst), "v"(g) : "memory");
    }
  }

  if (lane < 16) { sh_kj[lane] = idx_kj[row0 + lane]; sh_ji[lane] = idx_ji[row0 + lane]; }

  // stage sbf as zero-padded bf16 while the async copy is in flight
  for (int i = lane; i < 16 * 64; i += 32) {
    int r = i >> 6, k = i & 63;
    sh_s[i] = (k < SBFD) ? f2bf(sbf[(size_t)(row0 + r) * SBFD + k]) : (unsigned short)0;
  }
  for (int i = lane; i < 16 * 32; i += 32) { sh_mt[i] = 0; sh_ms[i] = 0; }

  asm volatile("s_wait_asynccnt 0x0" ::: "memory");
  __syncthreads();

  // convert t tile to zero-padded bf16
  for (int r = 0; r < 16; ++r)
    for (int k = lane; k < TTP; k += 32)
      sh_t[r * TTP + k] = (k < TTD) ? f2bf(sh_traw[r * TTD + k]) : (unsigned short)0;
  __syncthreads();

  // mid_t = t_tile @ W_t1 (padded)  : 10 WMMA, N-tile 0 only
  {
    F32x8 c;
#pragma unroll
    for (int g = 0; g < 8; ++g) c.f[g] = 0.f;
#pragma unroll
    for (int kt = 0; kt < 10; ++kt) {
      v16bf a = afrag_lds(sh_t, TTP, kt * 32, lane);
      c.v = wmma_bf16(a, bfrag(wf + WF_T1P, 1, kt, 0, lane), c.v);
    }
#pragma unroll
    for (int g = 0; g < 8; ++g) sh_mt[(g + hi * 8) * 32 + n] = f2bf(c.f[g]);
  }
  // mid_s = sbf_tile @ W_sbf1 (padded) : 2 WMMA
  {
    F32x8 c;
#pragma unroll
    for (int g = 0; g < 8; ++g) c.f[g] = 0.f;
#pragma unroll
    for (int kt = 0; kt < 2; ++kt) {
      v16bf a = afrag_lds(sh_s, 64, kt * 32, lane);
      c.v = wmma_bf16(a, bfrag(wf + WF_S1P, 1, kt, 0, lane), c.v);
    }
#pragma unroll
    for (int g = 0; g < 8; ++g) sh_ms[(g + hi * 8) * 32 + n] = f2bf(c.f[g]);
  }
  __syncthreads();

  // t_h = mid_t @ W_t2 ; s_h = mid_s @ W_sbf2  (K padded 8->32) : 4+4 WMMA
  v16bf amt = afrag_lds(sh_mt, 32, 0, lane);
  v16bf ams = afrag_lds(sh_ms, 32, 0, lane);
  F32x8 cth[4], csh[4];
#pragma unroll
  for (int nt = 0; nt < 4; ++nt) {
#pragma unroll
    for (int g = 0; g < 8; ++g) { cth[nt].f[g] = 0.f; csh[nt].f[g] = 0.f; }
    cth[nt].v = wmma_bf16(amt, bfrag(wf + WF_T2P, 4, 0, nt, lane), cth[nt].v);
    csh[nt].v = wmma_bf16(ams, bfrag(wf + WF_S2P, 4, 0, nt, lane), csh[nt].v);
  }

  // gather x_kj_down, fuse multiplies, scatter-add segment sum
#pragma unroll
  for (int nt = 0; nt < 4; ++nt) {
    int col = nt * 16 + n;
#pragma unroll
    for (int g = 0; g < 8; ++g) {
      int r = g + hi * 8;
      int kj = sh_kj[r], ji = sh_ji[r];
      float x = xkjd[(size_t)kj * Id + col];
      atomicAdd(&acc[(size_t)ji * Id + col], x * csh[nt].f[g] * cth[nt].f[g]);
    }
  }
}

// ---- residual layer: h += silu(silu(h@W1+b1)@W2+b2), h kept f32 in C-layout regs ----
__device__ __forceinline__ void res_layer(F32x8* Hc, unsigned short* sh_h, unsigned short* sh_m,
                                          const unsigned short* W1f, const float* b1,
                                          const unsigned short* W2f, const float* b2, int lane) {
  int n = lane & 15, hi = lane >> 4;
  v16bf A[4];
#pragma unroll
  for (int kt = 0; kt < 4; ++kt) A[kt] = afrag_lds(sh_h, Hd, kt * 32, lane);
  for (int nt = 0; nt < 8; ++nt) {
    int col = nt * 16 + n;
    float b = b1[col];
    F32x8 c;
#pragma unroll
    for (int g = 0; g < 8; ++g) c.f[g] = b;
#pragma unroll
    for (int kt = 0; kt < 4; ++kt) c.v = wmma_bf16(A[kt], bfrag(W1f, 8, kt, nt, lane), c.v);
#pragma unroll
    for (int g = 0; g < 8; ++g) sh_m[(g + hi * 8) * Hd + col] = f2bf(silu(c.f[g]));
  }
  __syncthreads();
#pragma unroll
  for (int kt = 0; kt < 4; ++kt) A[kt] = afrag_lds(sh_m, Hd, kt * 32, lane);
  for (int nt = 0; nt < 8; ++nt) {
    int col = nt * 16 + n;
    float b = b2[col];
    F32x8 c;
#pragma unroll
    for (int g = 0; g < 8; ++g) c.f[g] = b;
#pragma unroll
    for (int kt = 0; kt < 4; ++kt) c.v = wmma_bf16(A[kt], bfrag(W2f, 8, kt, nt, lane), c.v);
#pragma unroll
    for (int g = 0; g < 8; ++g) {
      float h = Hc[nt].f[g] + silu(c.f[g]);
      Hc[nt].f[g] = h;
      sh_h[(g + hi * 8) * Hd + col] = f2bf(h);
    }
  }
  __syncthreads();
}

// ================= kernel 4: edge phase 2 (up-proj, residual stacks, outputs) =================
__global__ void kedge2(const float* __restrict__ e1, const float* __restrict__ rbf,
                       const float* __restrict__ accp, const float* __restrict__ xji,
                       const unsigned short* __restrict__ wf,
                       const float* __restrict__ bb1, const float* __restrict__ bb2,
                       const float* __restrict__ b_lin,
                       const float* __restrict__ ba1, const float* __restrict__ ba2,
                       const float* __restrict__ W_rbf, float* __restrict__ out) {
  __shared__ __align__(16) unsigned short sh_h[16 * Hd];
  __shared__ __align__(16) unsigned short sh_m[16 * Hd];
  __shared__ float sh_rbf[16 * 8];
  int lane = threadIdx.x;
  int row0 = blockIdx.x * 16;
  int n = lane & 15, hi = lane >> 4;

  if (lane < 16) {
#pragma unroll
    for (int q = 0; q < NRr; ++q) sh_rbf[lane * 8 + q] = rbf[(size_t)(row0 + lane) * NRr + q];
  }

  // h = silu(acc @ W_up) + x_ji
  v16bf AU[2];
#pragma unroll
  for (int kt = 0; kt < 2; ++kt) AU[kt] = afrag_f32(accp + (size_t)row0 * Id, Id, kt * 32, lane);
  F32x8 Hc[8];
  for (int nt = 0; nt < 8; ++nt) {
    int col = nt * 16 + n;
    F32x8 c;
#pragma unroll
    for (int g = 0; g < 8; ++g) c.f[g] = 0.f;
#pragma unroll
    for (int kt = 0; kt < 2; ++kt) c.v = wmma_bf16(AU[kt], bfrag(wf + WF_UP, 8, kt, nt, lane), c.v);
#pragma unroll
    for (int g = 0; g < 8; ++g) {
      int r = g + hi * 8;
      float h = silu(c.f[g]) + xji[(size_t)(row0 + r) * Hd + col];
      Hc[nt].f[g] = h;
      sh_h[r * Hd + col] = f2bf(h);
    }
  }
  __syncthreads();

  // residual stack before skip (1 layer)
  res_layer(Hc, sh_h, sh_m, wf + WF_B1, bb1, wf + WF_B2, bb2, lane);

  // h = silu(h @ W_lin + b_lin) + e1
  {
    v16bf A[4];
#pragma unroll
    for (int kt = 0; kt < 4; ++kt) A[kt] = afrag_lds(sh_h, Hd, kt * 32, lane);
    for (int nt = 0; nt < 8; ++nt) {
      int col = nt * 16 + n;
      float b = b_lin[col];
      F32x8 c;
#pragma unroll
      for (int g = 0; g < 8; ++g) c.f[g] = b;
#pragma unroll
      for (int kt = 0; kt < 4; ++kt) c.v = wmma_bf16(A[kt], bfrag(wf + WF_LIN, 8, kt, nt, lane), c.v);
#pragma unroll
      for (int g = 0; g < 8; ++g) {
        int r = g + hi * 8;
        float h = silu(c.f[g]) + e1[(size_t)(row0 + r) * Hd + col];
        Hc[nt].f[g] = h;
        sh_h[r * Hd + col] = f2bf(h);
      }
    }
    __syncthreads();
  }

  // residual stack after skip (2 layers)
  res_layer(Hc, sh_h, sh_m, wf + WF_A1_0, ba1, wf + WF_A2_0, ba2, lane);
  res_layer(Hc, sh_h, sh_m, wf + WF_A1_1, ba1 + Hd, wf + WF_A2_1, ba2 + Hd, lane);

  // outputs: e1_out, e2_out = (rbf @ W_rbf) * e1_out
  for (int nt = 0; nt < 8; ++nt) {
    int col = nt * 16 + n;
#pragma unroll
    for (int g = 0; g < 8; ++g) {
      int r = g + hi * 8;
      float h = Hc[nt].f[g];
      out[(size_t)(row0 + r) * Hd + col] = h;
      float rd = 0.f;
#pragma unroll
      for (int q = 0; q < NRr; ++q) rd += sh_rbf[r * 8 + q] * W_rbf[q * Hd + col];
      out[(size_t)E_N * Hd + (size_t)(row0 + r) * Hd + col] = rd * h;
    }
  }
}

// ================= host launch =================
extern "C" void kernel_launch(void* const* d_in, const int* in_sizes, int n_in,
                              void* d_out, int out_size, void* d_ws, size_t ws_size,
                              hipStream_t stream) {
  const float* e1     = (const float*)d_in[0];
  const float* rbf    = (const float*)d_in[1];
  const float* sbf    = (const float*)d_in[2];
  const float* t      = (const float*)d_in[3];
  const int*   idx_kj = (const int*)d_in[4];
  const int*   idx_ji = (const int*)d_in[5];
  const float* W_rbf1 = (const float*)d_in[6];
  const float* W_rbf2 = (const float*)d_in[7];
  const float* W_sbf1 = (const float*)d_in[8];
  const float* W_sbf2 = (const float*)d_in[9];
  const float* W_t1   = (const float*)d_in[10];
  const float* W_t2   = (const float*)d_in[11];
  const float* W_rbf  = (const float*)d_in[12];
  const float* W_kj   = (const float*)d_in[13];
  const float* b_kj   = (const float*)d_in[14];
  const float* W_ji   = (const float*)d_in[15];
  const float* b_ji   = (const float*)d_in[16];
  const float* W_down = (const float*)d_in[17];
  const float* W_up   = (const float*)d_in[18];
  const float* Wb1    = (const float*)d_in[19];
  const float* bb1    = (const float*)d_in[20];
  const float* Wb2    = (const float*)d_in[21];
  const float* bb2    = (const float*)d_in[22];
  const float* W_lin  = (const float*)d_in[23];
  const float* b_lin  = (const float*)d_in[24];
  const float* Wa1    = (const float*)d_in[25];
  const float* ba1    = (const float*)d_in[26];
  const float* Wa2    = (const float*)d_in[27];
  const float* ba2    = (const float*)d_in[28];

  // workspace layout
  float* ws_xji  = (float*)d_ws;                       // [E,128] f32
  float* ws_xkjd = ws_xji + (size_t)E_N * Hd;          // [E,64]  f32
  float* ws_acc  = ws_xkjd + (size_t)E_N * Id;         // [E,64]  f32
  unsigned short* wf = (unsigned short*)(ws_acc + (size_t)E_N * Id);  // bf16 frag weights
  float* ws_wr   = (float*)(wf + WF_TOT);              // [6,128] f32

  kprep<<<(WF_TOT + NRr * Hd + 255) / 256, 256, 0, stream>>>(
      W_ji, W_kj, W_down, W_up, Wb1, Wb2, W_lin, Wa1, Wa2, W_rbf1, W_rbf2,
      W_sbf1, W_sbf2, W_t1, W_t2, wf, ws_wr);

  kzero<<<2048, 256, 0, stream>>>(ws_acc, (long long)E_N * Id);

  kedge1<<<E_N / 16, 32, 0, stream>>>(e1, rbf, b_ji, b_kj, wf, ws_wr, ws_xji, ws_xkjd);

  ktrip<<<T_N / 16, 32, 0, stream>>>(sbf, t, idx_kj, idx_ji, wf, ws_xkjd, ws_acc);

  kedge2<<<E_N / 16, 32, 0, stream>>>(e1, rbf, ws_acc, ws_xji, wf,
                                      bb1, bb2, b_lin, ba1, ba2, W_rbf, (float*)d_out);

  (void)in_sizes; (void)n_in; (void)out_size; (void)ws_size;
}
